// AttentionPool_14199161880847
// MI455X (gfx1250) — matrix operand order repaired and checked by reading, not compiled, not run on hardware
//
#include <hip/hip_runtime.h>
#include <hip/hip_bf16.h>

typedef __attribute__((ext_vector_type(16))) __bf16 v16bf;
typedef __attribute__((ext_vector_type(8)))  float  v8f;
typedef __attribute__((ext_vector_type(4)))  float  v4f;

#define NUM_GRAPHS 4096
#define IN_DIM 128
#define HID 256
#define EPS 1e-6f

// monotonic float <-> int encoding for atomicMax over signed floats
__device__ __forceinline__ int enc_f32(float f) {
    int i = __float_as_int(f);
    return (i < 0) ? (i ^ 0x7fffffff) : i;
}
__device__ __forceinline__ float dec_f32(int i) {
    return __int_as_float((i < 0) ? (i ^ 0x7fffffff) : i);
}
#define ENC_NEG_INF ((int)0x807FFFFF)   // enc(-inf)

__global__ void k_init(float* __restrict__ sumexp, int* __restrict__ maxenc) {
    int i = blockIdx.x * blockDim.x + threadIdx.x;
    if (i < NUM_GRAPHS) sumexp[i] = 0.0f;
    if (i == 0) *maxenc = ENC_NEG_INF;
}

// One wave32 computes logits for 16 nodes:
//   hidden(16x256) = h_tile(16x128) @ W1 (bf16 WMMA, inline-0 C seed)
//   logit = silu(hidden + b1) . W2 + b2  (bias folded into epilogue, shfl reduce)
__global__ __launch_bounds__(256) void k_gate(
        const float* __restrict__ h,  const float* __restrict__ W1,
        const float* __restrict__ b1, const float* __restrict__ W2,
        const float* __restrict__ b2, float* __restrict__ wlog,
        int* __restrict__ maxenc, int N)
{
    // W1 transposed to [n][k] in bf16, 16-element K-chunks XOR-swizzled by (n&7)
    // to spread the 256B row stride across LDS banks.  64 KB.
    __shared__ __align__(32) __bf16 w1t[HID * IN_DIM];

    const int tid = threadIdx.x;
    for (int idx = tid; idx < IN_DIM * HID; idx += 256) {
        int k = idx >> 8;                 // W1 is [128][256] row-major
        int n = idx & 255;
        int c = (k >> 4) ^ (n & 7);       // swizzled 16-elem chunk
        w1t[n * IN_DIM + (c << 4) + (k & 15)] = (__bf16)W1[idx];
    }
    __syncthreads();

    const int lane = tid & 31;
    const int lm   = lane & 15;           // column / row-in-tile selector
    const int lh   = lane >> 4;           // wave half
    const int tile = blockIdx.x * 8 + (tid >> 5);
    const int rowbase = tile * 16;

    // per-lane bias & gate vector for columns n = 16*j + lm
    float b1r[16], w2r[16];
    #pragma unroll
    for (int j = 0; j < 16; ++j) {
        b1r[j] = b1[j * 16 + lm];
        w2r[j] = W2[j * 16 + lm];
    }
    const float b2v = b2[0];

    // A fragments (contiguous k order, same as the B reads): row M = rowbase+lm (clamped),
// lane-half H: elems 0..15 -> K = 32*kc + 16H + e
    int arow = rowbase + lm;
    if (arow > N - 1) arow = N - 1;
    if (arow < 0) arow = 0;
    const float* hrow = h + (size_t)arow * IN_DIM;
    v16bf a[4];
    #pragma unroll
    for (int kc = 0; kc < 4; ++kc) {
        const int k0 = kc * 32 + 16 * lh;
        v4f p0 = *(const v4f*)(hrow + k0);
        v4f p1 = *(const v4f*)(hrow + k0 + 4);
        v4f p2 = *(const v4f*)(hrow + k0 + 8);
        v4f p3 = *(const v4f*)(hrow + k0 + 12);
        #pragma unroll
        for (int e = 0; e < 4; ++e) {
            a[kc][e]      = (__bf16)p0[e];
            a[kc][e + 4]  = (__bf16)p1[e];
            a[kc][e + 8]  = (__bf16)p2[e];
            a[kc][e + 12] = (__bf16)p3[e];
        }
    }

    float part[8] = {0.f, 0.f, 0.f, 0.f, 0.f, 0.f, 0.f, 0.f};

    #pragma unroll
    for (int j = 0; j < 16; ++j) {
        const int n = j * 16 + lm;        // hidden column for this lane
        // Hoist all four B fragments (contiguous 32B each in swizzled LDS)
        // ahead of the WMMA chain so ds_load latency overlaps prior VALU/WMMA.
        v16bf bfrag[4];
        #pragma unroll
        for (int kc = 0; kc < 4; ++kc) {
            const int ch = ((kc << 1) + lh) ^ (n & 7);
            bfrag[kc] = *(const v16bf*)&w1t[n * IN_DIM + (ch << 4)];
        }
        // zero accumulator -> backend can use WMMA's inline-0 SRC2
        v8f c = {0.f, 0.f, 0.f, 0.f, 0.f, 0.f, 0.f, 0.f};
        #pragma unroll
        for (int kc = 0; kc < 4; ++kc) {
            c = __builtin_amdgcn_wmma_f32_16x16x32_bf16(
                    false, a[kc], false, bfrag[kc], (short)0, c, false, false);
        }
        const float w2v = w2r[j];
        const float b1v = b1r[j];
        #pragma unroll
        for (int r = 0; r < 8; ++r) {     // silu + gate, C/D: row = r + 8*lh
            const float x = c[r] + b1v;                       // pk_add pairs
            const float sig = __builtin_amdgcn_rcpf(1.0f + __expf(-x));
            part[r] = fmaf(x * w2v, sig, part[r]);            // pk_mul + fmac
        }
    }

    // reduce the 16 column-partials across each wave half (xor<16 stays in half)
    #pragma unroll
    for (int r = 0; r < 8; ++r) {
        float v = part[r];
        v += __shfl_xor(v, 1, 32);
        v += __shfl_xor(v, 2, 32);
        v += __shfl_xor(v, 4, 32);
        v += __shfl_xor(v, 8, 32);
        part[r] = v + b2v;
    }

    if (lm == 0) {                        // lane 0 -> rows 0..7, lane 16 -> 8..15
        float m = -__builtin_inff();
        #pragma unroll
        for (int r = 0; r < 8; ++r) {
            const int row = rowbase + 8 * lh + r;
            if (row < N) {
                wlog[row] = part[r];
                m = fmaxf(m, part[r]);
            }
        }
        if (m > -3.0e38f) atomicMax(maxenc, enc_f32(m));
    }
}

__global__ void k_exp(float* __restrict__ w, const long long* __restrict__ batch,
                      float* __restrict__ sumexp, const int* __restrict__ maxenc,
                      int N)
{
    const int i = blockIdx.x * blockDim.x + threadIdx.x;
    if (i >= N) return;
    const float gmax = dec_f32(*maxenc);
    const float e = __expf(w[i] - gmax);
    w[i] = e;                                         // in-place: logits -> exp
    atomicAdd(&sumexp[(int)batch[i]], e);
}

// One block per graph; binary-search the sorted batch for the node range.
__global__ __launch_bounds__(128) void k_pool(
        const float* __restrict__ h, const long long* __restrict__ batch,
        const float* __restrict__ wexp, const float* __restrict__ sumexp,
        float* __restrict__ out, int N)
{
    const int g = blockIdx.x;
    const int d = threadIdx.x;
    __shared__ int s_lo, s_hi;
    if (d == 0) {
        int a = 0, b = N;
        while (a < b) { int m = (a + b) >> 1; if (batch[m] < (long long)g) a = m + 1; else b = m; }
        s_lo = a;
        b = N;
        while (a < b) { int m = (a + b) >> 1; if (batch[m] < (long long)(g + 1)) a = m + 1; else b = m; }
        s_hi = a;
    }
    __syncthreads();
    const int lo = s_lo, hi = s_hi;

    float acc = 0.0f;
    for (int n = lo; n < hi; ++n)
        acc += h[(size_t)n * IN_DIM + d] * wexp[n];

    out[g * IN_DIM + d] = acc / (sumexp[g] + EPS);
}

extern "C" void kernel_launch(void* const* d_in, const int* in_sizes, int n_in,
                              void* d_out, int out_size, void* d_ws, size_t ws_size,
                              hipStream_t stream)
{
    const float*     h     = (const float*)d_in[0];
    const long long* batch = (const long long*)d_in[1];   // jnp.int64
    const float*     W1    = (const float*)d_in[2];
    const float*     b1    = (const float*)d_in[3];
    const float*     W2    = (const float*)d_in[4];
    const float*     b2    = (const float*)d_in[5];
    float*           out   = (float*)d_out;
    const int        N     = in_sizes[1];

    float* wbuf   = (float*)d_ws;              // N floats: logits, then exp(w-max)
    float* sumexp = wbuf + N;                  // NUM_GRAPHS floats
    int*   maxenc = (int*)(sumexp + NUM_GRAPHS);

    k_init<<<(NUM_GRAPHS + 255) / 256, 256, 0, stream>>>(sumexp, maxenc);

    const int tiles   = (N + 15) / 16;         // 16 rows per wave
    const int blocksA = (tiles + 7) / 8;       // 8 waves per 256-thread block
    k_gate<<<blocksA, 256, 0, stream>>>(h, W1, b1, W2, b2, wbuf, maxenc, N);

    k_exp<<<(N + 255) / 256, 256, 0, stream>>>(wbuf, batch, sumexp, maxenc, N);

    k_pool<<<NUM_GRAPHS, 128, 0, stream>>>(h, batch, wbuf, sumexp, out, N);
}
